// HoloGraphBlockV7_39926015983649
// MI455X (gfx1250) — matrix operand order repaired and checked by reading, not compiled, not run on hardware
//
#include <hip/hip_runtime.h>
#include <hip/hip_bf16.h>

// ---------------- types ----------------
typedef _Float16 v16h __attribute__((ext_vector_type(16)));
typedef float    v8f  __attribute__((ext_vector_type(8)));

union FragH {
    v16h h;
    unsigned int u[8];
};

__device__ __forceinline__ unsigned short f2h(float f) {
    _Float16 h = (_Float16)f;
    unsigned short u;
    __builtin_memcpy(&u, &h, 2);
    return u;
}

// A-matrix 16x32 f16 fragment (ISA 7.12.2): vgpr g -> K = (g>=4?16:0) + hs*8 + (g&3)*2
__device__ __forceinline__ void load_afrag(FragH& f, const unsigned short* row, int k0, int hs) {
#pragma unroll
    for (int g = 0; g < 8; ++g) {
        int ka = ((g >= 4) ? 16 : 0) + hs * 8 + (g & 3) * 2;
        f.u[g] = *(const unsigned int*)(row + k0 + ka);
    }
}
// B-matrix 32x16 f16 fragment: same K order as A: vgpr g -> K = (g>=4?16:0) + hs*8 + (g&3)*2
__device__ __forceinline__ void load_bfrag(FragH& f, const unsigned short* row, int k0, int hs) {
#pragma unroll
    for (int g = 0; g < 8; ++g) {
        f.u[g] = *(const unsigned int*)(row + k0 + ((g >= 4) ? 16 : 0) + hs * 8 + (g & 3) * 2);
    }
}

// ---------------- constants ----------------
#define EB 512
#define HH 8
#define DD 64
#define BB 2
#define TT 1024
#define NT (BB * TT)          // 2048 tokens
#define HD (HH * DD)          // 512
#define SEG 16                // segments per (b,h) for the parallel scan
#define CL  (TT / SEG)        // 64 tokens per segment

// ---------------- weight pack: W (K x N, f32) -> Wt (N x K, f16) ----------------
__global__ void pack_wt_kernel(const float* __restrict__ src, unsigned short* __restrict__ dst,
                               int K, int N) {
    int idx = blockIdx.x * 256 + threadIdx.x;   // exact grid: N*K divisible by 256
    int n = idx / K;
    int k = idx - n * K;
    dst[idx] = f2h(src[(size_t)k * N + n]);
}

// ---------------- LayerNorm: row of 512, writes f32 + f16 ----------------
__global__ void ln_kernel(const float* __restrict__ x, const float* __restrict__ g,
                          const float* __restrict__ bt, float* __restrict__ out32,
                          unsigned short* __restrict__ out16) {
    int row = blockIdx.x;
    int tid = threadIdx.x;                      // 256 threads, 2 elems each
    const float* xr = x + (size_t)row * EB;
    float a0 = xr[tid], a1 = xr[tid + 256];
    __shared__ float s1[256], s2[256];
    s1[tid] = a0 + a1;
    s2[tid] = a0 * a0 + a1 * a1;
    __syncthreads();
    for (int st = 128; st > 0; st >>= 1) {
        if (tid < st) { s1[tid] += s1[tid + st]; s2[tid] += s2[tid + st]; }
        __syncthreads();
    }
    float mu  = s1[0] * (1.0f / EB);
    float var = s2[0] * (1.0f / EB) - mu * mu;
    float inv = rsqrtf(var + 1e-5f);
    size_t base = (size_t)row * EB;
    float y0 = (a0 - mu) * inv * g[tid]       + bt[tid];
    float y1 = (a1 - mu) * inv * g[tid + 256] + bt[tid + 256];
    out32[base + tid]       = y0;
    out32[base + tid + 256] = y1;
    out16[base + tid]       = f2h(y0);
    out16[base + tid + 256] = f2h(y1);
}

// ---------------- WMMA GEMM: A (M x K f16 rowmajor) x W (Wt stored N x K f16) ----------------
// Each wave computes a 32x64 output block: 2 A frags x 4 B frags -> 8 accumulators.
// Fragments are software-pipelined (next k-step loaded before current WMMAs issue)
// so the per-WMMA s_wait_loadcnt only drains the previous iteration's loads.
// Exact tiling everywhere -> EXEC all ones for every WMMA (ISA 7.12 requirement).
#define EPI_BIAS 0
#define EPI_WO   1
#define EPI_GELU 2
#define EPI_W2   3
#define EPI_PS   4

template <int EPI>
__global__ __launch_bounds__(256) void gemm_wmma(
        const unsigned short* __restrict__ A, const unsigned short* __restrict__ Wt,
        const float* __restrict__ bias, int M, int N, int K,
        float* __restrict__ out32, unsigned short* __restrict__ out16,
        const float* __restrict__ res1, const float* __restrict__ res2,
        float* __restrict__ out32b, const float* __restrict__ add1,
        const float* __restrict__ add2) {
    int wid  = blockIdx.x * 8 + (threadIdx.x >> 5);
    int lane = threadIdx.x & 31;
    int nstrip = N >> 6;                           // 64-wide strips in N
    int tm = (wid / nstrip) << 5;                  // 32-row strips in M
    int tn = (wid - (wid / nstrip) * nstrip) << 6;
    int ml = lane & 15;                            // M index within tile (A), N index (B/D)
    int hs = lane >> 4;                            // lane half select

    const unsigned short* Ar0 = A  + (size_t)(tm + ml) * K;
    const unsigned short* Ar1 = Ar0 + (size_t)16 * K;
    const unsigned short* Br0 = Wt + (size_t)(tn + ml) * K;  // Wt row n = W column n
    const unsigned short* Br1 = Br0 + (size_t)16 * K;
    const unsigned short* Br2 = Br0 + (size_t)32 * K;
    const unsigned short* Br3 = Br0 + (size_t)48 * K;

    v8f acc[8];
#pragma unroll
    for (int i = 0; i < 8; ++i) acc[i] = (v8f){};

    FragH af[2], bf[4];
    load_afrag(af[0], Ar0, 0, hs);
    load_afrag(af[1], Ar1, 0, hs);
    load_bfrag(bf[0], Br0, 0, hs);
    load_bfrag(bf[1], Br1, 0, hs);
    load_bfrag(bf[2], Br2, 0, hs);
    load_bfrag(bf[3], Br3, 0, hs);

    for (int k0 = 0; k0 < K - 32; k0 += 32) {
        FragH naf[2], nbf[4];
        load_afrag(naf[0], Ar0, k0 + 32, hs);
        load_afrag(naf[1], Ar1, k0 + 32, hs);
        load_bfrag(nbf[0], Br0, k0 + 32, hs);
        load_bfrag(nbf[1], Br1, k0 + 32, hs);
        load_bfrag(nbf[2], Br2, k0 + 32, hs);
        load_bfrag(nbf[3], Br3, k0 + 32, hs);
#pragma unroll
        for (int mi = 0; mi < 2; ++mi)
#pragma unroll
            for (int t = 0; t < 4; ++t)
                acc[mi * 4 + t] = __builtin_amdgcn_wmma_f32_16x16x32_f16(
                        false, af[mi].h, false, bf[t].h, (short)0,
                        acc[mi * 4 + t], false, false);
#pragma unroll
        for (int i = 0; i < 2; ++i) af[i] = naf[i];
#pragma unroll
        for (int i = 0; i < 4; ++i) bf[i] = nbf[i];
    }
#pragma unroll
    for (int mi = 0; mi < 2; ++mi)
#pragma unroll
        for (int t = 0; t < 4; ++t)
            acc[mi * 4 + t] = __builtin_amdgcn_wmma_f32_16x16x32_f16(
                    false, af[mi].h, false, bf[t].h, (short)0,
                    acc[mi * 4 + t], false, false);

#pragma unroll
    for (int mi = 0; mi < 2; ++mi) {
#pragma unroll
        for (int tt = 0; tt < 4; ++tt) {
#pragma unroll
            for (int r = 0; r < 8; ++r) {
                int m = tm + mi * 16 + r + (hs << 3);
                int n = tn + tt * 16 + ml;
                size_t idx = (size_t)m * N + n;
                float v = acc[mi * 4 + tt][r] + bias[n];
                if (EPI == EPI_BIAS) {
                    out32[idx] = v;
                } else if (EPI == EPI_WO) {
                    out32[idx] = res1[idx] + v + 0.1f * res2[idx];
                } else if (EPI == EPI_GELU) {
                    float gl = 0.5f * v * (1.0f + erff(v * 0.70710678118654752f));
                    out16[idx] = f2h(gl);
                } else if (EPI == EPI_W2) {
                    float x3 = res1[idx] + v;
                    out32[idx] = x3;
                    out16[idx] = f2h(x3);
                } else { // EPI_PS
                    out32[idx]  = add1[idx] + v;
                    out32b[idx] = add2[idx] + v;
                }
            }
        }
    }
}

// ---------------- small gates GEMM: x_n (2048x512) @ [Wd|Wgw|Wgf] (512x8 each) ----------------
__global__ void gates_kernel(const float* __restrict__ xn,
                             const float* __restrict__ Wd,  const float* __restrict__ bd,
                             const float* __restrict__ Wgw, const float* __restrict__ bgw,
                             const float* __restrict__ Wgf, const float* __restrict__ bgf,
                             float* __restrict__ gates) {
    int idx = blockIdx.x * 256 + threadIdx.x;   // 2048*24 exact
    int tok = idx / 24;
    int c   = idx - tok * 24;
    const float* W; const float* bb; int cc;
    if (c < 8)       { W = Wd;  bb = bd;  cc = c; }
    else if (c < 16) { W = Wgw; bb = bgw; cc = c - 8; }
    else             { W = Wgf; bb = bgf; cc = c - 16; }
    const float* xr = xn + (size_t)tok * EB;
    float s = bb[cc];
    for (int e = 0; e < EB; ++e) s += xr[e] * W[e * HH + cc];
    gates[(size_t)tok * 24 + c] = s;
}

// ---------------- per-(token,head) prep: l2norm k/q, tanh+gate v, decay ----------------
__global__ void prep_kernel(const float* __restrict__ qkv, const float* __restrict__ gates,
                            float* __restrict__ kb, float* __restrict__ qb,
                            float* __restrict__ vb, float* __restrict__ dec) {
    int w    = (blockIdx.x * 256 + threadIdx.x) >> 5;   // one wave per (token, head)
    int lane = threadIdx.x & 31;
    int tok = w >> 3, h = w & 7;
    const float* row = qkv + (size_t)tok * 1536;
    int hb = h * DD;
    float k0 = row[hb + lane],          k1 = row[hb + lane + 32];
    float q0 = row[512 + hb + lane],    q1 = row[512 + hb + lane + 32];
    float sk = k0 * k0 + k1 * k1;
    float sq = q0 * q0 + q1 * q1;
#pragma unroll
    for (int m = 16; m > 0; m >>= 1) {
        sk += __shfl_xor(sk, m, 32);
        sq += __shfl_xor(sq, m, 32);
    }
    float nk = fmaxf(sqrtf(sk), 1e-12f);
    float nq = fmaxf(sqrtf(sq), 1e-12f);
    const float* grow = gates + (size_t)tok * 24;
    float yd = grow[h];
    float d  = (yd > 20.0f) ? yd : log1pf(expf(yd));          // softplus
    float gw = 1.0f / (1.0f + expf(-grow[8 + h]));  gw = gw * gw;
    float gf = 1.0f / (1.0f + expf(-grow[16 + h])); gf = 1.0f - gf * gf;
    float scale = d * gw;
    float v0 = tanhf(row[1024 + hb + lane])      * scale;
    float v1 = tanhf(row[1024 + hb + lane + 32]) * scale;
    size_t ob = (size_t)tok * HD + hb;
    kb[ob + lane] = k0 / nk;  kb[ob + lane + 32] = k1 / nk;
    qb[ob + lane] = q0 / nq;  qb[ob + lane + 32] = q1 / nq;
    vb[ob + lane] = v0;       vb[ob + lane + 32] = v1;
    if (lane == 0) {
        float dcy = fminf(fmaxf(d * gf, 1e-6f), 0.999f);
        dec[tok * HH + h] = dcy;
    }
}

// ================= 3-pass segmented scan over T =================
// state_t = decay_t * state_{t-1} + v_t k_t^T   (scalar decay per (b,t,h))
// Pass 1: per (b,h,segment) local state S_p (from 0) and decay product Lambda_p.
// Pass 2: per (b,h) exclusive prefix across 16 segments (tiny); emits next_mem.
// Pass 3: per (b,h,segment) re-scan from prefix state; emits readouts.

// Pass 1: local segment state + decay product
__global__ __launch_bounds__(256) void scan_seg_state(
        const float* __restrict__ kb, const float* __restrict__ vb,
        const float* __restrict__ dec, float* __restrict__ segS,
        float* __restrict__ segL) {
    int blk = blockIdx.x;               // 0..255 : bh*SEG + p
    int bh  = blk >> 4;
    int p   = blk & 15;
    int b = bh >> 3, h = bh & 7;
    int tid = threadIdx.x;
    int row = tid >> 2;                 // i: 0..63
    int jb  = (tid & 3) * 16;
    float st[16];
#pragma unroll
    for (int u = 0; u < 16; ++u) st[u] = 0.0f;
    float lam = 1.0f;
    __shared__ float sk[DD], sv[DD];
    for (int t = p * CL; t < (p + 1) * CL; ++t) {
        int tok = b * TT + t;
        size_t base = (size_t)tok * HD + h * DD;
        if (tid < 64)       sk[tid]       = kb[base + tid];
        else if (tid < 128) sv[tid - 64]  = vb[base + tid - 64];
        __syncthreads();
        float dc = dec[tok * HH + h];
        lam *= dc;
        float vi = sv[row];
#pragma unroll
        for (int u = 0; u < 16; ++u) st[u] = st[u] * dc + vi * sk[jb + u];
        __syncthreads();
    }
    size_t sb = (size_t)blk * 4096 + (size_t)row * 64 + jb;
#pragma unroll
    for (int u = 0; u < 16; ++u) segS[sb + u] = st[u];
    if (tid == 0) segL[blk] = lam;
}

// Pass 2: sequential combine across segments (16 blocks, trivial)
__global__ __launch_bounds__(256) void scan_combine(
        const float* __restrict__ segS, const float* __restrict__ segL,
        float* __restrict__ segP, float* __restrict__ next_mem) {
    int bh  = blockIdx.x;               // 0..15
    int tid = threadIdx.x;              // 256 threads x 16 elems = 4096 state
    size_t eb = (size_t)tid * 16;
    float pre[16];
#pragma unroll
    for (int u = 0; u < 16; ++u) pre[u] = 0.0f;
    for (int p = 0; p < SEG; ++p) {
        size_t sb = (size_t)(bh * SEG + p) * 4096 + eb;
        float lam = segL[bh * SEG + p];
#pragma unroll
        for (int u = 0; u < 16; ++u) {
            segP[sb + u] = pre[u];                  // exclusive prefix state
            pre[u] = pre[u] * lam + segS[sb + u];
        }
    }
    size_t nb = (size_t)bh * 4096 + eb;             // (b,h,i,j) row-major
#pragma unroll
    for (int u = 0; u < 16; ++u) next_mem[nb + u] = pre[u];
}

// Pass 3: re-scan each segment from its prefix state, producing readouts
__global__ __launch_bounds__(256) void scan_readout(
        const float* __restrict__ kb, const float* __restrict__ qb,
        const float* __restrict__ vb, const float* __restrict__ dec,
        const float* __restrict__ segP, unsigned short* __restrict__ ro16) {
    int blk = blockIdx.x;               // 0..255
    int bh  = blk >> 4;
    int p   = blk & 15;
    int b = bh >> 3, h = bh & 7;
    int tid = threadIdx.x;
    int row = tid >> 2;
    int jb  = (tid & 3) * 16;
    float st[16];
    size_t sb = (size_t)blk * 4096 + (size_t)row * 64 + jb;
#pragma unroll
    for (int u = 0; u < 16; ++u) st[u] = segP[sb + u];
    __shared__ float sk[DD], sq[DD], sv[DD];
    for (int t = p * CL; t < (p + 1) * CL; ++t) {
        int tok = b * TT + t;
        size_t base = (size_t)tok * HD + h * DD;
        if (tid < 64)       sk[tid]       = kb[base + tid];
        else if (tid < 128) sq[tid - 64]  = qb[base + tid - 64];
        else if (tid < 192) sv[tid - 128] = vb[base + tid - 128];
        __syncthreads();
        float dc = dec[tok * HH + h];
        float vi = sv[row];
        float r = 0.0f;
#pragma unroll
        for (int u = 0; u < 16; ++u) {
            int j = jb + u;
            st[u] = st[u] * dc + vi * sk[j];
            r += st[u] * sq[j];
        }
        // reduce over the 4 consecutive lanes sharing row i (wave32-safe)
        r += __shfl_xor(r, 1, 32);
        r += __shfl_xor(r, 2, 32);
        if ((tid & 3) == 0) ro16[(size_t)tok * HD + h * DD + row] = f2h(r);
        __syncthreads();
    }
}

// ---------------- host launch ----------------
extern "C" void kernel_launch(void* const* d_in, const int* in_sizes, int n_in,
                              void* d_out, int out_size, void* d_ws, size_t ws_size,
                              hipStream_t stream) {
    (void)in_sizes; (void)n_in; (void)out_size; (void)ws_size;
    const float* x    = (const float*)d_in[0];
    const float* bmin = (const float*)d_in[1];
    const float* bmax = (const float*)d_in[2];
    const float* moti = (const float*)d_in[3];
    const float* Wk   = (const float*)d_in[4];   const float* bk  = (const float*)d_in[5];
    const float* Wq   = (const float*)d_in[6];   const float* bq  = (const float*)d_in[7];
    const float* Wv   = (const float*)d_in[8];   const float* bv  = (const float*)d_in[9];
    const float* Wo   = (const float*)d_in[10];  const float* bo  = (const float*)d_in[11];
    const float* Wd   = (const float*)d_in[12];  const float* bd  = (const float*)d_in[13];
    const float* Wgw  = (const float*)d_in[14];  const float* bgw = (const float*)d_in[15];
    const float* Wgf  = (const float*)d_in[16];  const float* bgf = (const float*)d_in[17];
    const float* Wps  = (const float*)d_in[18];  const float* bps = (const float*)d_in[19];
    const float* ln1g = (const float*)d_in[20];  const float* ln1b = (const float*)d_in[21];
    const float* ln2g = (const float*)d_in[22];  const float* ln2b = (const float*)d_in[23];
    const float* W1   = (const float*)d_in[24];  const float* b1  = (const float*)d_in[25];
    const float* W2   = (const float*)d_in[26];  const float* b2  = (const float*)d_in[27];

    char* ws = (char*)d_ws;
    size_t off = 0;
    auto alloc = [&](size_t bytes) -> char* {
        char* p = ws + off;
        off = (off + bytes + 255) & ~(size_t)255;
        return p;
    };

    unsigned short* wt_qkv = (unsigned short*)alloc((size_t)1536 * 512 * 2);
    unsigned short* wt_o   = (unsigned short*)alloc((size_t)512 * 512 * 2);
    unsigned short* wt_1   = (unsigned short*)alloc((size_t)2048 * 512 * 2);
    unsigned short* wt_2   = (unsigned short*)alloc((size_t)512 * 2048 * 2);
    unsigned short* wt_ps  = (unsigned short*)alloc((size_t)512 * 512 * 2);
    float*          bqkv   = (float*)alloc(1536 * 4);
    float*          xn32   = (float*)alloc((size_t)NT * EB * 4);
    unsigned short* xn16   = (unsigned short*)alloc((size_t)NT * EB * 2);
    float*          qkv32  = (float*)alloc((size_t)NT * 1536 * 4);
    float*          gates  = (float*)alloc((size_t)NT * 24 * 4);
    float*          kbuf   = (float*)alloc((size_t)NT * HD * 4);
    float*          qbuf   = (float*)alloc((size_t)NT * HD * 4);
    float*          vbuf   = (float*)alloc((size_t)NT * HD * 4);
    float*          decb   = (float*)alloc((size_t)NT * HH * 4);
    float*          segS   = (float*)alloc((size_t)BB * HH * SEG * 4096 * 4);  // 4 MB
    float*          segP   = (float*)alloc((size_t)BB * HH * SEG * 4096 * 4);  // 4 MB
    float*          segL   = (float*)alloc((size_t)BB * HH * SEG * 4);
    unsigned short* ro16   = (unsigned short*)alloc((size_t)NT * HD * 2);
    float*          x232   = (float*)alloc((size_t)NT * EB * 4);
    unsigned short* h16    = (unsigned short*)alloc((size_t)NT * EB * 2);
    unsigned short* ffn16  = (unsigned short*)alloc((size_t)NT * 2048 * 2);
    unsigned short* x316   = (unsigned short*)alloc((size_t)NT * EB * 2);

    float* out_x3   = (float*)d_out;                 // 2048*512
    float* out_mem  = out_x3 + (size_t)NT * EB;      // 2*8*64*64 = 65536
    float* out_minp = out_mem + 65536;               // 2048*512
    float* out_maxp = out_minp + (size_t)NT * EB;    // 2048*512

    // --- pack weights to transposed f16 ---
    pack_wt_kernel<<<1024, 256, 0, stream>>>(Wk, wt_qkv,               512, 512);
    pack_wt_kernel<<<1024, 256, 0, stream>>>(Wq, wt_qkv + 512 * 512,   512, 512);
    pack_wt_kernel<<<1024, 256, 0, stream>>>(Wv, wt_qkv + 1024 * 512,  512, 512);
    pack_wt_kernel<<<1024, 256, 0, stream>>>(Wo, wt_o,                 512, 512);
    pack_wt_kernel<<<4096, 256, 0, stream>>>(W1, wt_1,                 512, 2048);
    pack_wt_kernel<<<4096, 256, 0, stream>>>(W2, wt_2,                 2048, 512);
    pack_wt_kernel<<<1024, 256, 0, stream>>>(Wps, wt_ps,               512, 512);
    hipMemcpyAsync(bqkv,        bk, 512 * 4, hipMemcpyDeviceToDevice, stream);
    hipMemcpyAsync(bqkv + 512,  bq, 512 * 4, hipMemcpyDeviceToDevice, stream);
    hipMemcpyAsync(bqkv + 1024, bv, 512 * 4, hipMemcpyDeviceToDevice, stream);

    // --- LN1 ---
    ln_kernel<<<NT, 256, 0, stream>>>(x, ln1g, ln1b, xn32, xn16);

    // --- QKV GEMM (M=2048, N=1536, K=512): 64*24 = 1536 waves -> 192 blocks ---
    gemm_wmma<EPI_BIAS><<<192, 256, 0, stream>>>(xn16, wt_qkv, bqkv, NT, 1536, 512,
                                                 qkv32, nullptr, nullptr, nullptr,
                                                 nullptr, nullptr, nullptr);
    // --- gates (2048*24 dots) ---
    gates_kernel<<<192, 256, 0, stream>>>(xn32, Wd, bd, Wgw, bgw, Wgf, bgf, gates);

    // --- per-head prep (16384 waves / 8 = 2048 blocks) ---
    prep_kernel<<<2048, 256, 0, stream>>>(qkv32, gates, kbuf, qbuf, vbuf, decb);

    // --- 3-pass segmented scan ---
    scan_seg_state<<<BB * HH * SEG, 256, 0, stream>>>(kbuf, vbuf, decb, segS, segL);
    scan_combine  <<<BB * HH,       256, 0, stream>>>(segS, segL, segP, out_mem);
    scan_readout  <<<BB * HH * SEG, 256, 0, stream>>>(kbuf, qbuf, vbuf, decb, segP, ro16);

    // --- Wo GEMM + residual + motif (64*8 = 512 waves -> 64 blocks) ---
    gemm_wmma<EPI_WO><<<64, 256, 0, stream>>>(ro16, wt_o, bo, NT, 512, 512,
                                              x232, nullptr, x, moti,
                                              nullptr, nullptr, nullptr);
    // --- LN2 ---
    ln_kernel<<<NT, 256, 0, stream>>>(x232, ln2g, ln2b, xn32, h16);

    // --- W1 GEMM + exact GELU (M=2048,N=2048,K=512): 64*32 = 2048 waves -> 256 blocks ---
    gemm_wmma<EPI_GELU><<<256, 256, 0, stream>>>(h16, wt_1, b1, NT, 2048, 512,
                                                 nullptr, ffn16, nullptr, nullptr,
                                                 nullptr, nullptr, nullptr);
    // --- W2 GEMM + residual -> x3 (f32 out + f16 mirror): 64 blocks ---
    gemm_wmma<EPI_W2><<<64, 256, 0, stream>>>(ffn16, wt_2, b2, NT, 512, 2048,
                                              out_x3, x316, x232, nullptr,
                                              nullptr, nullptr, nullptr);
    // --- probe GEMM + dual fused adds: 64 blocks ---
    gemm_wmma<EPI_PS><<<64, 256, 0, stream>>>(x316, wt_ps, bps, NT, 512, 512,
                                              out_minp, nullptr, nullptr, nullptr,
                                              out_maxp, bmin, bmax);
}
